// AdvancedIITBlock_48086453846651
// MI455X (gfx1250) — compile-verified, matrix operand order checked by reading, not run on hardware
//
#include <hip/hip_runtime.h>
#include <cstddef>

// ---------------- CDNA5 WMMA types ----------------
typedef __attribute__((ext_vector_type(16))) __bf16 v16bf;
typedef __attribute__((ext_vector_type(8)))  float  v8f;

static constexpr int Dm   = 1024;   // model dim
static constexpr int Hh   = 16;     // heads
static constexpr int HD   = 64;     // head dim
static constexpr int Bsz  = 2;
static constexpr int Seq  = 2048;
static constexpr int Ntok = Bsz * Seq;   // 4096
static constexpr int Mem  = 128;
static constexpr int Dcat = 3 * Dm;      // 3072

// Convert 16 contiguous f32 (LDS) to a bf16 WMMA fragment.
__device__ __forceinline__ v16bf cvt16_f32_bf16(const float* __restrict__ p)
{
    v16bf r;
    #pragma unroll
    for (int i = 0; i < 16; ++i) r[i] = (__bf16)p[i];
    return r;
}

// gfx1250 async global->LDS copy, 16B per lane (tracked by ASYNCcnt)
__device__ __forceinline__ void async_ld_b128(unsigned lds_off, const float* gaddr)
{
    asm volatile("global_load_async_to_lds_b128 %0, %1, off"
                 :: "v"(lds_off), "v"(gaddr) : "memory");
}
__device__ __forceinline__ void wait_async0()
{
    asm volatile("s_wait_asynccnt 0x0" ::: "memory");
}

// =====================================================================
// Generic bf16 WMMA GEMM:  C = act(alpha * A@B + bias)
//   A: [M,K] f32 row-major (lda)   -- streamed via async copies, double-buffered
//   B: BNK==0 -> [K,N] f32 (ldb),  BNK==1 -> [N,K] f32 (ldb)  (i.e. A@B^T)
//   C: f32, row stride ldc, column offset col_off
// BM=128, BN=64, BK=32, 256 threads (8 waves), wave tile 32x32 (2x2 frags)
// =====================================================================
#define GBM 128
#define GBN 64
#define GBK 32

template <int ACT, int BNK>
__global__ __launch_bounds__(256) void gemm_bf16_wmma(
    const float* __restrict__ A, const float* __restrict__ Bm,
    const float* __restrict__ bias, float* __restrict__ C,
    int M, int N, int K, int lda, int ldb, int ldc, int col_off, float alpha)
{
    __shared__ alignas(16) float  Af32[2][GBM * GBK];   // 2 x 16KB (async target)
    __shared__ alignas(32) __bf16 Blds[2][GBN * GBK];   // 2 x 4KB, [n][k]

    const int t    = threadIdx.x;
    const int bm   = blockIdx.x;
    const int bn   = blockIdx.y;
    const int w    = t >> 5;
    const int lane = t & 31;
    const int l15  = lane & 15;
    const int half = (lane >> 4) & 1;
    const int wm   = w >> 1;   // 0..3
    const int wn   = w & 1;    // 0..1

    v8f acc[2][2] = {};
    const float* Abase = A + (size_t)bm * GBM * lda;

    // issue async copies of one 128x32 f32 A tile (4 x b128 per thread)
    auto issueA = [&](int buf, int k0) {
        unsigned lbase = (unsigned)(size_t)&Af32[buf][0];
        #pragma unroll
        for (int i = 0; i < 4; ++i) {
            int idx = t + 256 * i;            // 1024 x b128
            int row = idx >> 3;
            int c4  = (idx & 7) * 4;
            async_ld_b128(lbase + (unsigned)(row * GBK + c4) * 4u,
                          Abase + (size_t)row * lda + k0 + c4);
        }
    };
    // stage one B tile (transpose + f32->bf16) through the regular path
    auto stageB = [&](int buf, int k0) {
        if (BNK == 0) {
            #pragma unroll
            for (int i = 0; i < 2; ++i) {
                int idx = t + 256 * i;            // 512 float4
                int kr  = idx >> 4;               // 0..31
                int c4  = (idx & 15) * 4;         // 0..60
                const float4 v = *(const float4*)(Bm + (size_t)(k0 + kr) * ldb + bn * GBN + c4);
                __bf16* bl = &Blds[buf][0];
                bl[(c4 + 0) * GBK + kr] = (__bf16)v.x;
                bl[(c4 + 1) * GBK + kr] = (__bf16)v.y;
                bl[(c4 + 2) * GBK + kr] = (__bf16)v.z;
                bl[(c4 + 3) * GBK + kr] = (__bf16)v.w;
            }
        } else {
            #pragma unroll
            for (int i = 0; i < 2; ++i) {
                int idx = t + 256 * i;            // 512 float4
                int row = idx >> 3;               // 0..63
                int c4  = (idx & 7) * 4;
                const float4 v = *(const float4*)(Bm + (size_t)(bn * GBN + row) * ldb + k0 + c4);
                __bf16* d = &Blds[buf][row * GBK + c4];
                d[0] = (__bf16)v.x; d[1] = (__bf16)v.y; d[2] = (__bf16)v.z; d[3] = (__bf16)v.w;
            }
        }
    };

    const int nk = K / GBK;
    issueA(0, 0);
    stageB(0, 0);
    wait_async0();
    __syncthreads();

    for (int kt = 0; kt < nk; ++kt) {
        const int cur = kt & 1;
        if (kt + 1 < nk) issueA(1 - cur, (kt + 1) * GBK);   // overlap with compute

        // ---- fragments + WMMA on current buffers ----
        v16bf af[2], bf[2];
        #pragma unroll
        for (int fi = 0; fi < 2; ++fi)
            af[fi] = cvt16_f32_bf16(&Af32[cur][(wm * 32 + fi * 16 + l15) * GBK + half * 16]);
        #pragma unroll
        for (int fj = 0; fj < 2; ++fj)
            bf[fj] = *(const v16bf*)&Blds[cur][(wn * 32 + fj * 16 + l15) * GBK + half * 16];
        #pragma unroll
        for (int fi = 0; fi < 2; ++fi)
            #pragma unroll
            for (int fj = 0; fj < 2; ++fj)
                acc[fi][fj] = __builtin_amdgcn_wmma_f32_16x16x32_bf16(
                    false, af[fi], false, bf[fj], (short)0, acc[fi][fj], false, false);

        if (kt + 1 < nk) {
            stageB(1 - cur, (kt + 1) * GBK);   // overlaps the in-flight async A
            wait_async0();
            __syncthreads();
        }
    }

    // ---- epilogue: alpha, bias, activation, store f32 ----
    #pragma unroll
    for (int fi = 0; fi < 2; ++fi) {
        int mrow = bm * GBM + wm * 32 + fi * 16 + half * 8;
        #pragma unroll
        for (int fj = 0; fj < 2; ++fj) {
            int ncol = bn * GBN + wn * 32 + fj * 16 + l15;
            float bv = bias ? bias[ncol] : 0.0f;
            #pragma unroll
            for (int r = 0; r < 8; ++r) {
                float v = alpha * acc[fi][fj][r] + bv;
                if (ACT == 1)      v = fmaxf(v, 0.0f);
                else if (ACT == 2) v = 0.5f * v * (1.0f + erff(v * 0.70710678118f));
                C[(size_t)(mrow + r) * ldc + col_off + ncol] = v;
            }
        }
    }
}

// =====================================================================
// Flash attention (bf16 WMMA, online softmax).  hd=64, no mask.
// grid: (gl/64, H, B), block 128 (4 waves; wave = 16 query rows)
// Q/K/V/O: [B*gl, 1024] f32, head h occupies columns h*64..h*64+63
// K tile is streamed via async global->LDS; V needs a transpose so it
// goes through the VALU convert path (overlapping the async transfer).
// =====================================================================
__global__ __launch_bounds__(128) void flash_attn_wmma(
    const float* __restrict__ Q, const float* __restrict__ Kp,
    const float* __restrict__ Vp, float* __restrict__ O,
    int gl, float scale)
{
    __shared__ alignas(32) __bf16 Qlds[64 * 64];      // [q][hd]
    __shared__ alignas(16) float  Kf32[32 * 64];      // [key][hd] (async target)
    __shared__ alignas(32) __bf16 Vtlds[64 * 32];     // [hd][key]
    __shared__ alignas(32) __bf16 Plds[4 * 16 * 32];  // per-wave [q][key]

    const int t    = threadIdx.x;
    const int w    = t >> 5;
    const int lane = t & 31;
    const int l15  = lane & 15;
    const int half = (lane >> 4) & 1;
    const int qb   = blockIdx.x;
    const int h    = blockIdx.y;
    const int b    = blockIdx.z;
    const size_t rowbase = (size_t)b * gl;
    const int coff = h * HD;

    // ---- load Q block 64x64 (once) ----
    #pragma unroll
    for (int i = 0; i < 8; ++i) {
        int idx = t + 128 * i;       // 1024 float4
        int row = idx >> 4;
        int c4  = (idx & 15) * 4;
        float4 v = *(const float4*)(Q + (rowbase + (size_t)qb * 64 + row) * Dm + coff + c4);
        __bf16* d = &Qlds[row * 64 + c4];
        d[0] = (__bf16)v.x; d[1] = (__bf16)v.y; d[2] = (__bf16)v.z; d[3] = (__bf16)v.w;
    }
    __syncthreads();

    v16bf qf[2];
    #pragma unroll
    for (int s = 0; s < 2; ++s)
        qf[s] = *(const v16bf*)&Qlds[(w * 16 + l15) * 64 + s * 32 + half * 16];

    float mrow[8], lrow[8];
    v8f o[4] = {};
    #pragma unroll
    for (int r = 0; r < 8; ++r) { mrow[r] = -1e30f; lrow[r] = 0.0f; }

    const unsigned kbase = (unsigned)(size_t)&Kf32[0];
    const int nkb = gl / 32;
    for (int kb = 0; kb < nkb; ++kb) {
        // ---- async K tile (32x64 f32, 4 x b128 per thread) ----
        #pragma unroll
        for (int i = 0; i < 4; ++i) {
            int idx = t + 128 * i;    // 512 b128
            int row = idx >> 4;       // 0..31
            int c4  = (idx & 15) * 4;
            async_ld_b128(kbase + (unsigned)(row * 64 + c4) * 4u,
                          Kp + (rowbase + (size_t)kb * 32 + row) * Dm + coff + c4);
        }
        // ---- V^T (transpose + cvt) overlaps the async transfer ----
        #pragma unroll
        for (int i = 0; i < 4; ++i) {
            int idx = t + 128 * i;    // 512 float4
            int row = idx >> 4;       // 0..31
            int c4  = (idx & 15) * 4;
            float4 vv = *(const float4*)(Vp + (rowbase + (size_t)kb * 32 + row) * Dm + coff + c4);
            Vtlds[(c4 + 0) * 32 + row] = (__bf16)vv.x;
            Vtlds[(c4 + 1) * 32 + row] = (__bf16)vv.y;
            Vtlds[(c4 + 2) * 32 + row] = (__bf16)vv.z;
            Vtlds[(c4 + 3) * 32 + row] = (__bf16)vv.w;
        }
        wait_async0();
        __syncthreads();

        // ---- S = scale * Q @ K^T  (two 16-key fragments) ----
        v8f sfr[2];
        #pragma unroll
        for (int fj = 0; fj < 2; ++fj) {
            v8f sa = {};
            #pragma unroll
            for (int s = 0; s < 2; ++s) {
                v16bf kf = cvt16_f32_bf16(&Kf32[(fj * 16 + l15) * 64 + s * 32 + half * 16]);
                sa = __builtin_amdgcn_wmma_f32_16x16x32_bf16(
                        false, qf[s], false, kf, (short)0, sa, false, false);
            }
            sfr[fj] = sa * scale;
        }

        // ---- online softmax (rows live in one 16-lane half) ----
        float mnew[8], corr[8];
        #pragma unroll
        for (int r = 0; r < 8; ++r) {
            float mx = fmaxf(sfr[0][r], sfr[1][r]);
            #pragma unroll
            for (int m = 1; m < 16; m <<= 1) mx = fmaxf(mx, __shfl_xor(mx, m, 32));
            mnew[r] = fmaxf(mrow[r], mx);
            corr[r] = __expf(mrow[r] - mnew[r]);
        }
        #pragma unroll
        for (int r = 0; r < 8; ++r) {
            float p0 = __expf(sfr[0][r] - mnew[r]);
            float p1 = __expf(sfr[1][r] - mnew[r]);
            sfr[0][r] = p0; sfr[1][r] = p1;
            float s2 = p0 + p1;
            #pragma unroll
            for (int m = 1; m < 16; m <<= 1) s2 += __shfl_xor(s2, m, 32);
            lrow[r] = lrow[r] * corr[r] + s2;
            mrow[r] = mnew[r];
        }
        #pragma unroll
        for (int j = 0; j < 4; ++j)
            #pragma unroll
            for (int r = 0; r < 8; ++r) o[j][r] *= corr[r];

        // ---- P through per-wave LDS: C-layout -> A-fragment layout ----
        __bf16* pw = &Plds[w * 16 * 32];
        #pragma unroll
        for (int fj = 0; fj < 2; ++fj)
            #pragma unroll
            for (int r = 0; r < 8; ++r)
                pw[(r + 8 * half) * 32 + fj * 16 + l15] = (__bf16)sfr[fj][r];
        __builtin_amdgcn_wave_barrier();
        v16bf pf = *(const v16bf*)&pw[l15 * 32 + half * 16];

        // ---- O += P @ V ----
        #pragma unroll
        for (int j = 0; j < 4; ++j) {
            v16bf vf = *(const v16bf*)&Vtlds[(j * 16 + l15) * 32 + half * 16];
            o[j] = __builtin_amdgcn_wmma_f32_16x16x32_bf16(
                      false, pf, false, vf, (short)0, o[j], false, false);
        }
        __syncthreads();
    }

    // ---- epilogue ----
    #pragma unroll
    for (int j = 0; j < 4; ++j)
        #pragma unroll
        for (int r = 0; r < 8; ++r) {
            size_t row = rowbase + (size_t)qb * 64 + w * 16 + half * 8 + r;
            O[row * Dm + coff + j * 16 + l15] = o[j][r] / lrow[r];
        }
}

// =====================================================================
// Small f32 helper kernels
// =====================================================================
__global__ void mean_tokens_k(const float* __restrict__ x, float* __restrict__ ctx, int S_, int Dd)
{
    int idx = blockIdx.x * blockDim.x + threadIdx.x;     // b*D + d
    if (idx >= Bsz * Dd) return;
    int b = idx / Dd, d = idx % Dd;
    const float* p = x + ((size_t)b * S_) * Dd + d;
    float s = 0.0f;
    for (int i = 0; i < S_; ++i) s += p[(size_t)i * Dd];
    ctx[idx] = s / (float)S_;
}

__global__ __launch_bounds__(512) void ctx_mlp_k(
    const float* __restrict__ ctx, const float* __restrict__ w1, const float* __restrict__ b1,
    const float* __restrict__ w2, const float* __restrict__ b2, float* __restrict__ adapt)
{
    __shared__ float cv[1024];
    __shared__ float hv[512];
    int b = blockIdx.x, t = threadIdx.x;
    cv[t]       = ctx[b * 1024 + t];
    cv[t + 512] = ctx[b * 1024 + t + 512];
    __syncthreads();
    float acc = b1[t];
    for (int k = 0; k < 1024; ++k) acc += cv[k] * w1[k * 512 + t];
    hv[t] = fmaxf(acc, 0.0f);
    __syncthreads();
    for (int j = 0; j < 2; ++j) {
        int d = t + 512 * j;
        float a2 = b2[d];
        for (int k = 0; k < 512; ++k) a2 += hv[k] * w2[k * 1024 + d];
        adapt[b * 1024 + d] = a2;
    }
}

__global__ void add_pos_k(const float* __restrict__ x, const float* __restrict__ pos,
                          const float* __restrict__ adapt, float* __restrict__ out, size_t n)
{
    size_t i = (size_t)blockIdx.x * blockDim.x + threadIdx.x;
    if (i >= n) return;
    int d = (int)(i % Dm);
    size_t sd = i / Dm;
    int s = (int)(sd % Seq);
    int b = (int)(sd / Seq);
    out[i] = x[i] + pos[(size_t)s * Dm + d] * (1.0f + adapt[b * Dm + d]);
}

__global__ void softmax_rows_k(float* __restrict__ X, int rows, int cols)
{
    int r = blockIdx.x * blockDim.x + threadIdx.x;
    if (r >= rows) return;
    float* p = X + (size_t)r * cols;
    float mx = -1e30f;
    for (int i = 0; i < cols; ++i) mx = fmaxf(mx, p[i]);
    float s = 0.0f;
    for (int i = 0; i < cols; ++i) { float e = __expf(p[i] - mx); p[i] = e; s += e; }
    float inv = 1.0f / s;
    for (int i = 0; i < cols; ++i) p[i] *= inv;
}

// out = LN(fa*A + fb*B) * g + be      (B, sa, sb may be null)
__global__ __launch_bounds__(256) void layernorm_res_k(
    const float* __restrict__ A, const float* __restrict__ Bres,
    const float* __restrict__ sa, const float* __restrict__ sb,
    const float* __restrict__ g, const float* __restrict__ be,
    float* __restrict__ out)
{
    __shared__ float red[256];
    int row = blockIdx.x, t = threadIdx.x;
    float fa = sa ? sa[0] : 1.0f;
    float fb = sb ? sb[0] : 1.0f;
    const float* a = A + (size_t)row * Dm;
    const float* b = Bres ? (Bres + (size_t)row * Dm) : nullptr;
    float vals[4];
    float sum = 0.0f;
    #pragma unroll
    for (int i = 0; i < 4; ++i) {
        int c = t + 256 * i;
        float v = fa * a[c] + (b ? fb * b[c] : 0.0f);
        vals[i] = v; sum += v;
    }
    red[t] = sum; __syncthreads();
    for (int st = 128; st > 0; st >>= 1) { if (t < st) red[t] += red[t + st]; __syncthreads(); }
    float mu = red[0] / (float)Dm;
    __syncthreads();
    float vs = 0.0f;
    #pragma unroll
    for (int i = 0; i < 4; ++i) { float d = vals[i] - mu; vs += d * d; }
    red[t] = vs; __syncthreads();
    for (int st = 128; st > 0; st >>= 1) { if (t < st) red[t] += red[t + st]; __syncthreads(); }
    float inv = rsqrtf(red[0] / (float)Dm + 1e-5f);
    #pragma unroll
    for (int i = 0; i < 4; ++i) {
        int c = t + 256 * i;
        out[(size_t)row * Dm + c] = (vals[i] - mu) * inv * g[c] + be[c];
    }
}

__global__ void pool_mean_k(const float* __restrict__ in, float* __restrict__ out,
                            int gl, int scale, size_t n)
{
    size_t i = (size_t)blockIdx.x * blockDim.x + threadIdx.x;   // over B*gl*D
    if (i >= n) return;
    int d = (int)(i % Dm);
    size_t gd = i / Dm;
    int g = (int)(gd % gl);
    int b = (int)(gd / gl);
    const float* p = in + ((size_t)b * gl * scale + (size_t)g * scale) * Dm + d;
    float s = 0.0f;
    for (int j = 0; j < scale; ++j) s += p[(size_t)j * Dm];
    out[i] = s / (float)scale;
}

__global__ void interp_to_cat_k(const float* __restrict__ in, float* __restrict__ cat,
                                int gl, int coloff, size_t n)
{
    size_t i = (size_t)blockIdx.x * blockDim.x + threadIdx.x;   // over B*S*D
    if (i >= n) return;
    int d = (int)(i % Dm);
    size_t sd = i / Dm;
    int s = (int)(sd % Seq);
    int b = (int)(sd / Seq);
    float pos = ((float)s + 0.5f) * ((float)gl / (float)Seq) - 0.5f;
    pos = fminf(fmaxf(pos, 0.0f), (float)(gl - 1));
    int lo = (int)floorf(pos);
    int hi = min(lo + 1, gl - 1);
    float w = pos - (float)lo;
    const float* base = in + (size_t)b * gl * Dm;
    float v = base[(size_t)lo * Dm + d] * (1.0f - w) + base[(size_t)hi * Dm + d] * w;
    cat[((size_t)b * Seq + s) * Dcat + coloff + d] = v;
}

__global__ void gate_k(const float* __restrict__ gh, const float* __restrict__ w2,
                       const float* __restrict__ b2, float* __restrict__ gates, int rows)
{
    int r = blockIdx.x * blockDim.x + threadIdx.x;
    if (r >= rows) return;
    const float* p = gh + (size_t)r * 256;
    float a0 = b2[0], a1 = b2[1];
    for (int k = 0; k < 256; ++k) { float v = p[k]; a0 += v * w2[k * 2]; a1 += v * w2[k * 2 + 1]; }
    float m = fmaxf(a0, a1);
    float e0 = __expf(a0 - m), e1 = __expf(a1 - m);
    float inv = 1.0f / (e0 + e1);
    gates[r * 2] = e0 * inv; gates[r * 2 + 1] = e1 * inv;
}

__global__ void gated_combine_k(const float* __restrict__ b1, const float* __restrict__ b2,
                                const float* __restrict__ gates, float* __restrict__ out, size_t n)
{
    size_t i = (size_t)blockIdx.x * blockDim.x + threadIdx.x;
    if (i >= n) return;
    size_t r = i / Dm;
    out[i] = gates[r * 2] * b1[i] + gates[r * 2 + 1] * b2[i];
}

// =====================================================================
// Host side
// =====================================================================
static inline void launch_gemm(hipStream_t st, int act, int bnk,
                               const float* A, const float* Bm, const float* bias, float* C,
                               int M, int N, int K, int lda, int ldb, int ldc,
                               int coloff = 0, float alpha = 1.0f)
{
    dim3 g(M / GBM, N / GBN), b(256);
    if (bnk == 0) {
        if (act == 0)      gemm_bf16_wmma<0,0><<<g,b,0,st>>>(A,Bm,bias,C,M,N,K,lda,ldb,ldc,coloff,alpha);
        else if (act == 1) gemm_bf16_wmma<1,0><<<g,b,0,st>>>(A,Bm,bias,C,M,N,K,lda,ldb,ldc,coloff,alpha);
        else               gemm_bf16_wmma<2,0><<<g,b,0,st>>>(A,Bm,bias,C,M,N,K,lda,ldb,ldc,coloff,alpha);
    } else {
        gemm_bf16_wmma<0,1><<<g,b,0,st>>>(A,Bm,bias,C,M,N,K,lda,ldb,ldc,coloff,alpha);
    }
}

// input indices (setup_inputs insertion order: x, then params)
enum {
    I_X = 0, I_POS,
    I_CTX_W1, I_CTX_B1, I_CTX_W2, I_CTX_B2,
    I_MEMBANK,
    I_MQ_W, I_MQ_B, I_MK_W, I_MK_B, I_MV_W, I_MV_B, I_MO_W, I_MO_B,
    I_MEMLN_G, I_MEMLN_B,
    I_L0_QW, I_L0_QB, I_L0_KW, I_L0_KB, I_L0_VW, I_L0_VB, I_L0_OW, I_L0_OB,
    I_L1_QW, I_L1_QB, I_L1_KW, I_L1_KB, I_L1_VW, I_L1_VB, I_L1_OW, I_L1_OB,
    I_L2_QW, I_L2_QB, I_L2_KW, I_L2_KB, I_L2_VW, I_L2_VB, I_L2_OW, I_L2_OB,
    I_FUS_W, I_FUS_B,
    I_HALN_G, I_HALN_B,
    I_F1_W1, I_F1_B1, I_F1_W2, I_F1_B2,
    I_F2_W1, I_F2_B1, I_F2_W2, I_F2_B2,
    I_G_W1, I_G_B1, I_G_W2, I_G_B2,
    I_FFNLN_G, I_FFNLN_B,
    I_PHI, I_MI,
    I_FINLN_G, I_FINLN_B
};

extern "C" void kernel_launch(void* const* d_in, const int* in_sizes, int n_in,
                              void* d_out, int out_size, void* d_ws, size_t ws_size,
                              hipStream_t stream)
{
    (void)in_sizes; (void)n_in; (void)out_size; (void)ws_size;
    auto in = [&](int i) { return (const float*)d_in[i]; };

    float* ws = (float*)d_ws;
    size_t off = 0;
    auto alloc = [&](size_t n) { float* p = ws + off; off += n; return p; };

    const size_t ND = (size_t)Ntok * Dm;   // 4M floats
    float* ctx    = alloc(Bsz * Dm);
    float* adapt  = alloc(Bsz * Dm);
    float* xpe    = alloc(ND);
    float* qbuf   = alloc(ND);
    float* mkbuf  = alloc((size_t)Mem * Dm);
    float* mvbuf  = alloc((size_t)Mem * Dm);
    float* scores = alloc((size_t)Ntok * Mem);
    float* attm   = alloc(ND);
    float* moproj = alloc(ND);
    float* xmem   = alloc(ND);
    float* xs1    = alloc((size_t)Bsz * 1024 * Dm);
    float* xs2    = alloc((size_t)Bsz * 512 * Dm);
    float* lq     = alloc(ND);
    float* lk     = alloc(ND);
    float* lv     = alloc(ND);
    float* latt   = alloc(ND);
    float* lout   = alloc(ND);
    float* catb   = alloc((size_t)Ntok * Dcat);
    float* fused  = alloc(ND);
    float* xatt   = alloc(ND);
    float* gh     = alloc((size_t)Ntok * 256);
    float* gates  = alloc((size_t)Ntok * 2);
    float* h1     = alloc((size_t)Ntok * 2048);
    float* b1b    = alloc(ND);
    float* h2     = alloc((size_t)Ntok * 4096);
    float* b2b    = alloc(ND);
    float* ffn    = alloc(ND);
    float* xffn   = alloc(ND);

    const size_t BSD = ND;
    const int EB = 256;
    const int ew_blocks = (int)((BSD + EB - 1) / EB);

    // 1) adaptive positional encoding
    mean_tokens_k<<<(Bsz * Dm + EB - 1) / EB, EB, 0, stream>>>(in(I_X), ctx, Seq, Dm);
    ctx_mlp_k<<<Bsz, 512, 0, stream>>>(ctx, in(I_CTX_W1), in(I_CTX_B1), in(I_CTX_W2), in(I_CTX_B2), adapt);
    add_pos_k<<<ew_blocks, EB, 0, stream>>>(in(I_X), in(I_POS), adapt, xpe, BSD);

    // 2) memory-bank cross attention
    launch_gemm(stream, 0, 0, xpe,           in(I_MQ_W), in(I_MQ_B), qbuf,  Ntok, Dm, Dm, Dm, Dm, Dm);
    launch_gemm(stream, 0, 0, in(I_MEMBANK), in(I_MK_W), in(I_MK_B), mkbuf, Mem,  Dm, Dm, Dm, Dm, Dm);
    launch_gemm(stream, 0, 0, in(I_MEMBANK), in(I_MV_W), in(I_MV_B), mvbuf, Mem,  Dm, Dm, Dm, Dm, Dm);
    // scores = (q @ mk^T) / sqrt(D)   (B in [N,K] mode)
    launch_gemm(stream, 0, 1, qbuf, mkbuf, nullptr, scores, Ntok, Mem, Dm, Dm, Dm, Mem, 0, 0.03125f);
    softmax_rows_k<<<(Ntok + EB - 1) / EB, EB, 0, stream>>>(scores, Ntok, Mem);
    launch_gemm(stream, 0, 0, scores, mvbuf, nullptr, attm, Ntok, Dm, Mem, Mem, Dm, Dm);
    launch_gemm(stream, 0, 0, attm, in(I_MO_W), in(I_MO_B), moproj, Ntok, Dm, Dm, Dm, Dm, Dm);
    layernorm_res_k<<<Ntok, 256, 0, stream>>>(moproj, xpe, nullptr, nullptr,
                                              in(I_MEMLN_G), in(I_MEMLN_B), xmem);

    // 3) hierarchical attention, levels 0..2
    const int qw[3] = {I_L0_QW, I_L1_QW, I_L2_QW};
    for (int l = 0; l < 3; ++l) {
        int scale = 1 << l;
        int gl = Seq / scale;
        int rows = Bsz * gl;
        const float* xs;
        if (l == 0) xs = xmem;
        else {
            float* dst = (l == 1) ? xs1 : xs2;
            size_t n = (size_t)rows * Dm;
            pool_mean_k<<<(int)((n + EB - 1) / EB), EB, 0, stream>>>(xmem, dst, gl, scale, n);
            xs = dst;
        }
        launch_gemm(stream, 0, 0, xs, in(qw[l] + 0), in(qw[l] + 1), lq, rows, Dm, Dm, Dm, Dm, Dm);
        launch_gemm(stream, 0, 0, xs, in(qw[l] + 2), in(qw[l] + 3), lk, rows, Dm, Dm, Dm, Dm, Dm);
        launch_gemm(stream, 0, 0, xs, in(qw[l] + 4), in(qw[l] + 5), lv, rows, Dm, Dm, Dm, Dm, Dm);
        dim3 fg(gl / 64, Hh, Bsz);
        flash_attn_wmma<<<fg, 128, 0, stream>>>(lq, lk, lv, latt, gl, 0.125f);
        if (l == 0) {
            // project straight into concat buffer (ldc=3072, coloff=0)
            launch_gemm(stream, 0, 0, latt, in(qw[l] + 6), in(qw[l] + 7), catb,
                        rows, Dm, Dm, Dm, Dm, Dcat, 0);
        } else {
            launch_gemm(stream, 0, 0, latt, in(qw[l] + 6), in(qw[l] + 7), lout,
                        rows, Dm, Dm, Dm, Dm, Dm);
            interp_to_cat_k<<<ew_blocks, EB, 0, stream>>>(lout, catb, gl, l * Dm, BSD);
        }
    }

    // 4) fusion + LN
    launch_gemm(stream, 0, 0, catb, in(I_FUS_W), in(I_FUS_B), fused, Ntok, Dm, Dcat, Dcat, Dm, Dm);
    layernorm_res_k<<<Ntok, 256, 0, stream>>>(fused, xmem, nullptr, nullptr,
                                              in(I_HALN_G), in(I_HALN_B), xatt);

    // 5) gated dual-branch FFN
    launch_gemm(stream, 1, 0, xatt, in(I_G_W1), in(I_G_B1), gh, Ntok, 256, Dm, Dm, 256, 256);
    gate_k<<<(Ntok + EB - 1) / EB, EB, 0, stream>>>(gh, in(I_G_W2), in(I_G_B2), gates, Ntok);
    launch_gemm(stream, 2, 0, xatt, in(I_F1_W1), in(I_F1_B1), h1, Ntok, 2048, Dm, Dm, 2048, 2048);
    launch_gemm(stream, 0, 0, h1, in(I_F1_W2), in(I_F1_B2), b1b, Ntok, Dm, 2048, 2048, Dm, Dm);
    launch_gemm(stream, 2, 0, xatt, in(I_F2_W1), in(I_F2_B1), h2, Ntok, 4096, Dm, Dm, 4096, 4096);
    launch_gemm(stream, 0, 0, h2, in(I_F2_W2), in(I_F2_B2), b2b, Ntok, Dm, 4096, 4096, Dm, Dm);
    gated_combine_k<<<ew_blocks, EB, 0, stream>>>(b1b, b2b, gates, ffn, BSD);
    layernorm_res_k<<<Ntok, 256, 0, stream>>>(ffn, xatt, nullptr, nullptr,
                                              in(I_FFNLN_G), in(I_FFNLN_B), xffn);

    // 6) final combine + LN -> d_out
    layernorm_res_k<<<Ntok, 256, 0, stream>>>(xffn, xmem, in(I_PHI), in(I_MI),
                                              in(I_FINLN_G), in(I_FINLN_B), (float*)d_out);
}
